// TransformerBlock_17729624998447
// MI455X (gfx1250) — matrix operand order repaired and checked by reading, not compiled, not run on hardware
//
#include <hip/hip_runtime.h>
#include <hip/hip_bf16.h>

#define BDIM 2
#define SEQ  2048
#define DM   1024
#define NH   16
#define DH   64
#define HID  4096
#define TOK  (BDIM*SEQ)

typedef __attribute__((ext_vector_type(16))) __bf16 v16bf;
typedef __attribute__((ext_vector_type(8)))  __bf16 v8bf;
typedef __attribute__((ext_vector_type(8)))  float  v8f;

static __device__ __forceinline__ v16bf ld16(const __bf16* p0, const __bf16* p1) {
    v8bf lo = *(const v8bf*)p0;
    v8bf hi = *(const v8bf*)p1;
    v16bf r;
#pragma unroll
    for (int i = 0; i < 8; ++i) { r[i] = lo[i]; r[8 + i] = hi[i]; }
    return r;
}

static __device__ __forceinline__ v8f wmma_bf16(v16bf a, v16bf b, v8f c) {
    return __builtin_amdgcn_wmma_f32_16x16x32_bf16(false, a, false, b, (short)0, c, false, false);
}

// ---------------- weight convert + transpose: f32 [K][N] -> bf16 [N][K] -------------
__global__ void convT_k(const float* __restrict__ W, __bf16* __restrict__ Wt, int K, int N) {
    size_t i = (size_t)blockIdx.x * 256 + threadIdx.x;
    if (i >= (size_t)K * N) return;
    int k = (int)(i / N), n = (int)(i % N);
    Wt[(size_t)n * K + k] = (__bf16)W[i];
}

// ---------------- layernorm: f32 row -> bf16 row (x-mu)*rsqrt(var+eps)*w ------------
__global__ void ln_k(const float* __restrict__ X, const float* __restrict__ w,
                     __bf16* __restrict__ out) {
    int row = blockIdx.x;
    int t = threadIdx.x;
    const float* x = X + (size_t)row * DM;
    float v[4]; float s = 0.f, ss = 0.f;
#pragma unroll
    for (int i = 0; i < 4; ++i) { float a = x[t + i * 256]; v[i] = a; s += a; ss += a * a; }
    __shared__ float r1[256], r2[256];
    r1[t] = s; r2[t] = ss; __syncthreads();
    for (int off = 128; off > 0; off >>= 1) {
        if (t < off) { r1[t] += r1[t + off]; r2[t] += r2[t + off]; }
        __syncthreads();
    }
    float mu  = r1[0] * (1.0f / DM);
    float var = r2[0] * (1.0f / DM) - mu * mu;
    float rs  = rsqrtf(var + 1e-5f);
    __bf16* orow = out + (size_t)row * DM;
#pragma unroll
    for (int i = 0; i < 4; ++i) { int c = t + i * 256; orow[c] = (__bf16)((v[i] - mu) * rs * w[c]); }
}

// ---------------- generic WMMA GEMM: C = A(MxK, bf16) * Bt(NxK, bf16)^T --------------
// Block tile 128x64, 8 waves, 32x32 per wave (2 A frags x 2 B frags -> 4 WMMAs/step).
// EPI 0: outF[row*N+col] = acc (+ resid)        (f32)
// EPI 2: scatter qkv columns into Q, K (s x d) and Vt (d x s), bf16
template <int EPI>
__global__ void __launch_bounds__(256)
gemm_k(const __bf16* __restrict__ A, const __bf16* __restrict__ Bt, int N, int K,
       const float* __restrict__ resid, float* __restrict__ outF,
       __bf16* __restrict__ Qo, __bf16* __restrict__ Ko, __bf16* __restrict__ Vto) {
    const int lane = threadIdx.x & 31;
    const int wave = threadIdx.x >> 5;
    const int ln = lane & 15, kh = lane >> 4;
    const int m0 = blockIdx.y * 128 + (wave >> 1) * 32;
    const int c0 = blockIdx.x * 64 + (wave & 1) * 32;
    const __bf16* a0row = A  + (size_t)(m0 + ln) * K;
    const __bf16* a1row = A  + (size_t)(m0 + 16 + ln) * K;
    const __bf16* b0row = Bt + (size_t)(c0 + ln) * K;
    const __bf16* b1row = Bt + (size_t)(c0 + 16 + ln) * K;
    v8f acc00 = {0, 0, 0, 0, 0, 0, 0, 0};
    v8f acc01 = acc00, acc10 = acc00, acc11 = acc00;
    for (int k = 0; k < K; k += 32) {
        // WGP-scope prefetch: pull into all cache levels (locality=3)
        __builtin_prefetch(a0row + k + 256, 0, 3);
        __builtin_prefetch(a1row + k + 256, 0, 3);
        __builtin_prefetch(b0row + k + 256, 0, 3);
        __builtin_prefetch(b1row + k + 256, 0, 3);
        v16bf a0 = ld16(a0row + k + kh * 8,  a0row + k + 16 + kh * 8);
        v16bf a1 = ld16(a1row + k + kh * 8,  a1row + k + 16 + kh * 8);
        v16bf b0 = ld16(b0row + k + kh * 8, b0row + k + 16 + kh * 8);
        v16bf b1 = ld16(b1row + k + kh * 8, b1row + k + 16 + kh * 8);
        acc00 = wmma_bf16(a0, b0, acc00);
        acc01 = wmma_bf16(a0, b1, acc01);
        acc10 = wmma_bf16(a1, b0, acc10);
        acc11 = wmma_bf16(a1, b1, acc11);
    }
    const int rb0 = m0 + kh * 8;
    const int rb1 = m0 + 16 + kh * 8;
    if (EPI == 0) {
        auto storeF = [&](const v8f& acc, int rbase, int cc) {
#pragma unroll
            for (int r = 0; r < 8; ++r) {
                size_t i = (size_t)(rbase + r) * N + cc + ln;
                float v = acc[r];
                if (resid) v += resid[i];
                outF[i] = v;
            }
        };
        storeF(acc00, rb0, c0);
        storeF(acc01, rb0, c0 + 16);
        storeF(acc10, rb1, c0);
        storeF(acc11, rb1, c0 + 16);
    } else { // EPI == 2 : qkv scatter
        auto scatter = [&](const v8f& acc, int rbase, int cc) {
#pragma unroll
            for (int r = 0; r < 8; ++r) {
                int tok = rbase + r;
                int b = tok / SEQ, s = tok % SEQ;
                int c = cc + ln;
                int head = c / (DH * 3);
                int rem  = c % (DH * 3);
                int d    = rem / 3;
                int comp = rem % 3;
                __bf16 hv = (__bf16)acc[r];
                if (comp == 0)      Qo[(((size_t)(b * NH + head)) * SEQ + s) * DH + d] = hv;
                else if (comp == 1) Ko[(((size_t)(b * NH + head)) * SEQ + s) * DH + d] = hv;
                else                Vto[(((size_t)(b * NH + head)) * DH + d) * SEQ + s] = hv;
            }
        };
        scatter(acc00, rb0, c0);
        scatter(acc01, rb0, c0 + 16);
        scatter(acc10, rb1, c0);
        scatter(acc11, rb1, c0 + 16);
    }
}

// ---------------- fused FC1 + gated GELU: m = (h@W_a) * gelu(h@W_g), bf16 ------------
// Block tile 128x64 of m; per wave 32 rows x 32 cols for both a- and g-branches.
__global__ void __launch_bounds__(256)
gemm_mlp1_k(const __bf16* __restrict__ A, const __bf16* __restrict__ Wt,
            __bf16* __restrict__ Mout) {
    const int lane = threadIdx.x & 31;
    const int wave = threadIdx.x >> 5;
    const int ln = lane & 15, kh = lane >> 4;
    const int m0 = blockIdx.y * 128 + (wave >> 1) * 32;
    const int c0 = blockIdx.x * 64 + (wave & 1) * 32;
    const __bf16* a0row = A + (size_t)(m0 + ln) * DM;
    const __bf16* a1row = A + (size_t)(m0 + 16 + ln) * DM;
    const __bf16* ba0   = Wt + (size_t)(c0 + ln) * DM;
    const __bf16* ba1   = Wt + (size_t)(c0 + 16 + ln) * DM;
    const __bf16* bg0   = Wt + (size_t)(HID + c0 + ln) * DM;
    const __bf16* bg1   = Wt + (size_t)(HID + c0 + 16 + ln) * DM;
    v8f aA00 = {0, 0, 0, 0, 0, 0, 0, 0};
    v8f aA01 = aA00, aA10 = aA00, aA11 = aA00;
    v8f aG00 = aA00, aG01 = aA00, aG10 = aA00, aG11 = aA00;
    for (int k = 0; k < DM; k += 32) {
        __builtin_prefetch(a0row + k + 256, 0, 3);
        __builtin_prefetch(a1row + k + 256, 0, 3);
        __builtin_prefetch(ba0 + k + 256, 0, 3);
        __builtin_prefetch(bg0 + k + 256, 0, 3);
        v16bf a0 = ld16(a0row + k + kh * 8,  a0row + k + 16 + kh * 8);
        v16bf a1 = ld16(a1row + k + kh * 8,  a1row + k + 16 + kh * 8);
        v16bf f0 = ld16(ba0 + k + kh * 8, ba0 + k + 16 + kh * 8);
        v16bf f1 = ld16(ba1 + k + kh * 8, ba1 + k + 16 + kh * 8);
        v16bf g0 = ld16(bg0 + k + kh * 8, bg0 + k + 16 + kh * 8);
        v16bf g1 = ld16(bg1 + k + kh * 8, bg1 + k + 16 + kh * 8);
        aA00 = wmma_bf16(a0, f0, aA00);
        aA01 = wmma_bf16(a0, f1, aA01);
        aA10 = wmma_bf16(a1, f0, aA10);
        aA11 = wmma_bf16(a1, f1, aA11);
        aG00 = wmma_bf16(a0, g0, aG00);
        aG01 = wmma_bf16(a0, g1, aG01);
        aG10 = wmma_bf16(a1, g0, aG10);
        aG11 = wmma_bf16(a1, g1, aG11);
    }
    const int rb0 = m0 + kh * 8;
    const int rb1 = m0 + 16 + kh * 8;
    auto storeG = [&](const v8f& aa, const v8f& gg, int rbase, int cc) {
#pragma unroll
        for (int r = 0; r < 8; ++r) {
            float a = aa[r], g = gg[r];
            float gl = 0.5f * g * (1.0f + tanhf(0.7978845608028654f * (g + 0.044715f * g * g * g)));
            Mout[(size_t)(rbase + r) * HID + cc + ln] = (__bf16)(a * gl);
        }
    };
    storeG(aA00, aG00, rb0, c0);
    storeG(aA01, aG01, rb0, c0 + 16);
    storeG(aA10, aG10, rb1, c0);
    storeG(aA11, aG11, rb1, c0 + 16);
}

// ---------------- flash attention: per (b,h), 64 q-rows per block, 4 waves -----------
__global__ void __launch_bounds__(128)
attn_k(const __bf16* __restrict__ Q, const __bf16* __restrict__ Km,
       const __bf16* __restrict__ Vt, __bf16* __restrict__ Aout) {
    const int lane = threadIdx.x & 31;
    const int wave = threadIdx.x >> 5;          // 0..3
    const int ln = lane & 15, kh = lane >> 4;
    const int b = blockIdx.z, h = blockIdx.y;
    const int q0 = blockIdx.x * 64 + wave * 16;
    const __bf16* Qb = Q  + ((size_t)(b * NH + h)) * SEQ * DH;
    const __bf16* Kb = Km + ((size_t)(b * NH + h)) * SEQ * DH;
    const __bf16* Vb = Vt + ((size_t)(b * NH + h)) * DH * SEQ;
    __shared__ __align__(16) __bf16 plds[4][16][32];   // per-wave P tile (16x32)

    const __bf16* qrow = Qb + (size_t)(q0 + ln) * DH;
    v16bf qa0 = ld16(qrow + kh * 8,      qrow + 16 + kh * 8);   // d 0..31
    v16bf qa1 = ld16(qrow + 32 + kh * 8, qrow + 48 + kh * 8);   // d 32..63

    v8f o0 = {0, 0, 0, 0, 0, 0, 0, 0};
    v8f o1 = o0, o2 = o0, o3 = o0;
    float mrow[8], lrow[8];
#pragma unroll
    for (int r = 0; r < 8; ++r) { mrow[r] = -3.0e38f; lrow[r] = 0.f; }

    for (int kb = 0; kb < SEQ; kb += 32) {
        const __bf16* k0r = Kb + (size_t)(kb + ln) * DH;
        const __bf16* k1r = Kb + (size_t)(kb + 16 + ln) * DH;
        __builtin_prefetch(Kb + (size_t)(kb + 32 + ln) * DH, 0, 3);
        v16bf bk00 = ld16(k0r + kh * 8,      k0r + 16 + kh * 8);
        v16bf bk01 = ld16(k0r + 32 + kh * 8, k0r + 48 + kh * 8);
        v16bf bk10 = ld16(k1r + kh * 8,      k1r + 16 + kh * 8);
        v16bf bk11 = ld16(k1r + 32 + kh * 8, k1r + 48 + kh * 8);
        v8f s0 = {0, 0, 0, 0, 0, 0, 0, 0};
        v8f s1 = s0;
        s0 = wmma_bf16(qa0, bk00, s0);
        s0 = wmma_bf16(qa1, bk01, s0);
        s1 = wmma_bf16(qa0, bk10, s1);
        s1 = wmma_bf16(qa1, bk11, s1);

        float p0[8], p1[8], alpha[8];
#pragma unroll
        for (int r = 0; r < 8; ++r) {
            float a = s0[r] * 0.125f, c = s1[r] * 0.125f;   // 1/sqrt(64)
            float lm = fmaxf(a, c);
#pragma unroll
            for (int msk = 1; msk < 16; msk <<= 1) lm = fmaxf(lm, __shfl_xor(lm, msk, 32));
            float mn = fmaxf(mrow[r], lm);
            float al = __expf(mrow[r] - mn);
            float e0 = __expf(a - mn), e1 = __expf(c - mn);
            float rs = e0 + e1;
#pragma unroll
            for (int msk = 1; msk < 16; msk <<= 1) rs += __shfl_xor(rs, msk, 32);
            lrow[r] = lrow[r] * al + rs;
            mrow[r] = mn;
            alpha[r] = al;
            p0[r] = e0; p1[r] = e1;
        }
#pragma unroll
        for (int r = 0; r < 8; ++r) {
            o0[r] *= alpha[r]; o1[r] *= alpha[r]; o2[r] *= alpha[r]; o3[r] *= alpha[r];
            int row = kh * 8 + r;
            plds[wave][row][ln]      = (__bf16)p0[r];
            plds[wave][row][16 + ln] = (__bf16)p1[r];
        }
        // re-shape P from C-fragment to A-fragment (wave-local LDS; DS ops in-order)
        v16bf pa = ld16(&plds[wave][ln][kh * 8], &plds[wave][ln][16 + kh * 8]);

        v16bf vf0 = ld16(Vb + (size_t)(ln) * SEQ + kb + kh * 8,      Vb + (size_t)(ln) * SEQ + kb + 16 + kh * 8);
        v16bf vf1 = ld16(Vb + (size_t)(16 + ln) * SEQ + kb + kh * 8, Vb + (size_t)(16 + ln) * SEQ + kb + 16 + kh * 8);
        v16bf vf2 = ld16(Vb + (size_t)(32 + ln) * SEQ + kb + kh * 8, Vb + (size_t)(32 + ln) * SEQ + kb + 16 + kh * 8);
        v16bf vf3 = ld16(Vb + (size_t)(48 + ln) * SEQ + kb + kh * 8, Vb + (size_t)(48 + ln) * SEQ + kb + 16 + kh * 8);
        o0 = wmma_bf16(pa, vf0, o0);
        o1 = wmma_bf16(pa, vf1, o1);
        o2 = wmma_bf16(pa, vf2, o2);
        o3 = wmma_bf16(pa, vf3, o3);
    }
#pragma unroll
    for (int r = 0; r < 8; ++r) {
        int row = q0 + kh * 8 + r;
        float inv = 1.0f / lrow[r];
        size_t base = ((size_t)(b * SEQ + row)) * DM + h * DH;
        Aout[base + ln]      = (__bf16)(o0[r] * inv);
        Aout[base + 16 + ln] = (__bf16)(o1[r] * inv);
        Aout[base + 32 + ln] = (__bf16)(o2[r] * inv);
        Aout[base + 48 + ln] = (__bf16)(o3[r] * inv);
    }
}

// ------------------------------------ launch -----------------------------------------
extern "C" void kernel_launch(void* const* d_in, const int* in_sizes, int n_in,
                              void* d_out, int out_size, void* d_ws, size_t ws_size,
                              hipStream_t stream) {
    (void)in_sizes; (void)n_in; (void)out_size; (void)ws_size;
    const float* X    = (const float*)d_in[0];
    const float* Wqkv = (const float*)d_in[1];
    const float* Wout = (const float*)d_in[2];
    const float* ln1w = (const float*)d_in[3];
    const float* Wfc1 = (const float*)d_in[4];
    const float* Wfc2 = (const float*)d_in[5];
    const float* ln2w = (const float*)d_in[6];
    float* outX = (float*)d_out;

    char* p = (char*)d_ws;
    auto carve = [&](size_t bytes) -> char* {
        char* r = p; p += (bytes + 255) & ~(size_t)255; return r;
    };
    __bf16* WqkvT = (__bf16*)carve((size_t)3072 * 1024 * 2);
    __bf16* WoutT = (__bf16*)carve((size_t)1024 * 1024 * 2);
    __bf16* Wfc1T = (__bf16*)carve((size_t)8192 * 1024 * 2);
    __bf16* Wfc2T = (__bf16*)carve((size_t)1024 * 4096 * 2);
    __bf16* h1    = (__bf16*)carve((size_t)TOK * DM * 2);
    __bf16* Qb    = (__bf16*)carve((size_t)BDIM * NH * SEQ * DH * 2);
    __bf16* Kbuf  = (__bf16*)carve((size_t)BDIM * NH * SEQ * DH * 2);
    __bf16* Vtb   = (__bf16*)carve((size_t)BDIM * NH * SEQ * DH * 2);
    __bf16* attno = (__bf16*)carve((size_t)TOK * DM * 2);
    float*  X1    = (float*) carve((size_t)TOK * DM * 4);
    __bf16* h2    = (__bf16*)carve((size_t)TOK * DM * 2);
    __bf16* mbuf  = (__bf16*)carve((size_t)TOK * HID * 2);

    // weight convert + transpose (bf16, [N][K])
    convT_k<<<(1024 * 3072 + 255) / 256, 256, 0, stream>>>(Wqkv, WqkvT, 1024, 3072);
    convT_k<<<(1024 * 1024 + 255) / 256, 256, 0, stream>>>(Wout, WoutT, 1024, 1024);
    convT_k<<<(1024 * 8192 + 255) / 256, 256, 0, stream>>>(Wfc1, Wfc1T, 1024, 8192);
    convT_k<<<(4096 * 1024 + 255) / 256, 256, 0, stream>>>(Wfc2, Wfc2T, 4096, 1024);

    // LN1
    ln_k<<<TOK, 256, 0, stream>>>(X, ln1w, h1);
    // QKV projection with scatter into Q / K / V^T
    gemm_k<2><<<dim3(3072 / 64, TOK / 128), 256, 0, stream>>>(
        h1, WqkvT, 3072, 1024, nullptr, nullptr, Qb, Kbuf, Vtb);
    // attention
    attn_k<<<dim3(SEQ / 64, NH, BDIM), 128, 0, stream>>>(Qb, Kbuf, Vtb, attno);
    // out projection + residual -> X1 (f32)
    gemm_k<0><<<dim3(DM / 64, TOK / 128), 256, 0, stream>>>(
        attno, WoutT, DM, DM, X, X1, nullptr, nullptr, nullptr);
    // LN2
    ln_k<<<TOK, 256, 0, stream>>>(X1, ln2w, h2);
    // fused FC1 + gated GELU
    gemm_mlp1_k<<<dim3(HID / 64, TOK / 128), 256, 0, stream>>>(h2, Wfc1T, mbuf);
    // FC2 + residual -> output (f32)
    gemm_k<0><<<dim3(DM / 64, TOK / 128), 256, 0, stream>>>(
        mbuf, Wfc2T, DM, HID, X1, outX, nullptr, nullptr, nullptr);
}
